// _SelectiveStateSpace_36867999268920
// MI455X (gfx1250) — matrix operand order repaired and checked by reading, not compiled, not run on hardware
//
#include <hip/hip_runtime.h>
#include <hip/hip_bf16.h>
#include <math.h>

// Problem constants (fixed by the reference).
#define BSZ   4
#define SEQ   4096
#define DM    1024
#define DI    2048
#define DS    16
#define DTR   64
#define NP    96                 // dt_rank + 2*d_state
#define MROWS (BSZ * SEQ)        // 16384

// Epilogue selectors for the fused GEMM.
#define EPI_NONE        0
#define EPI_SILU_GE     1        // v = silu(v) when col >= silu_from
#define EPI_SOFTPLUS2   2        // v = softplus(v + 2*bias[col])

// gfx1250 async global->LDS path (guarded; falls back to VGPR staging).
#if defined(__has_builtin)
#  if __has_builtin(__builtin_amdgcn_global_load_async_to_lds_b32) && \
      __has_builtin(__builtin_amdgcn_s_wait_asynccnt)
#    define USE_ASYNC_LDS 1
#  endif
#endif
#ifndef USE_ASYNC_LDS
#  define USE_ASYNC_LDS 0
#endif

#if USE_ASYNC_LDS
typedef __attribute__((address_space(1))) int g_i32;   // global
typedef __attribute__((address_space(3))) int l_i32;   // LDS
#endif

typedef __attribute__((ext_vector_type(16))) __bf16 v16bf;
typedef __attribute__((ext_vector_type(8)))  float  v8f;

__device__ __forceinline__ float silu_f(float x) { return x / (1.0f + __expf(-x)); }

__device__ __forceinline__ v8f wmma_bf16(const v16bf& a, const v16bf& b, const v8f& c) {
  return __builtin_amdgcn_wmma_f32_16x16x32_bf16(
      /*neg_a=*/false, a, /*neg_b=*/false, b,
      /*c_mod=*/(short)0, c, /*reuse_a=*/false, /*reuse_b=*/false);
}

__device__ __forceinline__ void store_tile(
    const v8f& acc, float* __restrict__ C, int ldc,
    int mt, int nt, int half, int lidx,
    int epi, int silu_from, const float* __restrict__ bias2)
{
  const int col = nt + lidx;
#pragma unroll
  for (int r = 0; r < 8; ++r) {
    float v = acc[r];
    if (epi == EPI_SILU_GE) {
      if (col >= silu_from) v = silu_f(v);
    } else if (epi == EPI_SOFTPLUS2) {
      v += 2.0f * bias2[col];
      v = (v > 20.0f) ? v : log1pf(__expf(v));
    }
    C[(size_t)(mt + 8 * half + r) * ldc + col] = v;
  }
}

// ---------------------------------------------------------------------------
// C[M,N] = A[M,K] (row-major, lda) * Bw[N,K]^T (row-major weights, ldb)
// fp32 in/out, bf16 WMMA with fp32 accumulation, fused epilogue.
// One wave computes a 32x32 output macro-tile (2x2 WMMA tiles): each A frag
// feeds 2 WMMAs and each B frag feeds 2 WMMAs (register-level reuse).
// Block = 8 waves -> 32(M) x 256(N) strip. K loop unrolled x2 (8 static WMMA).
// Fragment layouts per CDNA5 ISA §7.12.2:
//   A (16x32 bf16): half h=lane>>4, row M=lane&15, elem j -> K=j+8h+(j>=8?8:0)
//   B (32x16 bf16): col N=lane&15, elem j -> K=j+8h+(j>=8?8:0)
//   C/D (16x16 f32): col N=lane&15, vgpr r -> row M=r+8h
// ---------------------------------------------------------------------------
__global__ __launch_bounds__(256) void gemm_wmma_bf16(
    const float* __restrict__ A, int lda,
    const float* __restrict__ Bw, int ldb,
    float* __restrict__ C, int ldc,
    int N, int K,
    int epi, int silu_from, const float* __restrict__ bias2)
{
  const int wave = threadIdx.x >> 5;
  const int lane = threadIdx.x & 31;
  const int half = lane >> 4;
  const int lidx = lane & 15;
  const int m0 = blockIdx.x * 32;
  const int n0 = blockIdx.y * 256 + wave * 32;
  if (n0 >= N) return;                        // fully out-of-range wave

  const float* __restrict__ Ap0 = A + (size_t)(m0 + lidx) * lda;
  const float* __restrict__ Ap1 = A + (size_t)(m0 + 16 + lidx) * lda;
  const int nr0 = min(n0 + lidx, N - 1);      // clamp loads in-bounds
  const int nr1 = min(n0 + 16 + lidx, N - 1);
  const float* __restrict__ Bp0 = Bw + (size_t)nr0 * ldb;
  const float* __restrict__ Bp1 = Bw + (size_t)nr1 * ldb;

  v8f acc00 = {}, acc01 = {}, acc10 = {}, acc11 = {};
#pragma unroll 2
  for (int k0 = 0; k0 < K; k0 += 32) {
    if (k0 + 32 < K) {                        // prefetch next K tile
      __builtin_prefetch(Ap0 + k0 + 32, 0, 1);
      __builtin_prefetch(Ap1 + k0 + 32, 0, 1);
      __builtin_prefetch(Bp0 + k0 + 32, 0, 1);
      __builtin_prefetch(Bp1 + k0 + 32, 0, 1);
    }
    v16bf a0, a1, b0, b1;
#pragma unroll
    for (int j = 0; j < 16; ++j) {
      const int ka = k0 + j + 8 * half + ((j >= 8) ? 8 : 0);
      const int kb = k0 + j + 8 * half + ((j >= 8) ? 8 : 0);
      a0[j] = (__bf16)Ap0[ka];
      a1[j] = (__bf16)Ap1[ka];
      b0[j] = (__bf16)Bp0[kb];
      b1[j] = (__bf16)Bp1[kb];
    }
    acc00 = wmma_bf16(a0, b0, acc00);
    acc01 = wmma_bf16(a0, b1, acc01);
    acc10 = wmma_bf16(a1, b0, acc10);
    acc11 = wmma_bf16(a1, b1, acc11);
  }

  const bool n1ok = (n0 + 16) < N;
  store_tile(acc00, C, ldc, m0,      n0,      half, lidx, epi, silu_from, bias2);
  if (n1ok)
    store_tile(acc01, C, ldc, m0,      n0 + 16, half, lidx, epi, silu_from, bias2);
  store_tile(acc10, C, ldc, m0 + 16, n0,      half, lidx, epi, silu_from, bias2);
  if (n1ok)
    store_tile(acc11, C, ldc, m0 + 16, n0 + 16, half, lidx, epi, silu_from, bias2);
}

// ---------------------------------------------------------------------------
// Depthwise causal conv1d (width 4) + bias + SiLU.
// xz rows are [x_in(0..DI-1) | silu(z)(DI..2*DI-1)], row stride 2*DI.
// ---------------------------------------------------------------------------
__global__ __launch_bounds__(256) void conv_silu_k(
    const float* __restrict__ xz, const float* __restrict__ cw,
    const float* __restrict__ cb, float* __restrict__ u)
{
  const int idx = blockIdx.x * 256 + threadIdx.x;   // over MROWS*DI
  const int d = idx & (DI - 1);
  const int m = idx >> 11;                          // DI = 2048
  const int l = m & (SEQ - 1);
  const int b = m >> 12;                            // SEQ = 4096
  float acc = cb[d];
#pragma unroll
  for (int j = 0; j < 4; ++j) {
    const int ll = l - 3 + j;
    if (ll >= 0)
      acc += cw[d * 4 + j] * xz[(size_t)(b * SEQ + ll) * (2 * DI) + d];
  }
  u[idx] = silu_f(acc);
}

// ---------------------------------------------------------------------------
// Sequential selective scan over L. One thread per (batch, channel) chain,
// h[DS] in registers; per-step B_t/C_t (shared across all 256 channels of the
// block) staged into LDS — via GLOBAL_LOAD_ASYNC_TO_LDS_B32 (ASYNCcnt) when
// the toolchain exposes it, else via VGPR staging.
// y_ssm + u*D written into the (consumed) x_in half of xz.
// ---------------------------------------------------------------------------
__global__ __launch_bounds__(256) void scan_k(
    const float* __restrict__ delta, const float* __restrict__ u,
    const float* __restrict__ params, const float* __restrict__ A_log,
    const float* __restrict__ D_param, float* __restrict__ xz)
{
  __shared__ float sBC[2 * DS];                   // [B_t | C_t]
  const int b = blockIdx.x;                       // 0..BSZ-1
  const int d = blockIdx.y * 256 + threadIdx.x;   // 0..DI-1

  float Arow[DS];
#pragma unroll
  for (int n = 0; n < DS; ++n) Arow[n] = -__expf(A_log[d * DS + n]);
  const float Dp = D_param[d];

  float h[DS];
#pragma unroll
  for (int n = 0; n < DS; ++n) h[n] = 0.0f;

  for (int l = 0; l < SEQ; ++l) {
    const int m = b * SEQ + l;
    // Stage B_t (params cols 64..79) and C_t (cols 80..95) into LDS.
    if (threadIdx.x < 2 * DS) {                   // wave 0, all lanes active
      const int t = threadIdx.x;
      const float* src = params + (size_t)m * NP + DTR + t;
#if USE_ASYNC_LDS
      __builtin_amdgcn_global_load_async_to_lds_b32(
          (g_i32*)src, (l_i32*)&sBC[t],
          /*offset=*/0, /*cpol=*/0);
#else
      sBC[t] = *src;
#endif
    }
#if USE_ASYNC_LDS
    __builtin_amdgcn_s_wait_asynccnt(0);
#endif
    __syncthreads();

    const float dl = delta[(size_t)m * DI + d];
    const float ul = u[(size_t)m * DI + d];
    const float du = dl * ul;
    float y = 0.0f;
#pragma unroll
    for (int n = 0; n < DS; ++n) {
      const float dA = __expf(dl * Arow[n]);
      h[n] = dA * h[n] + du * sBC[n];
      y += h[n] * sBC[DS + n];
    }
    y += ul * Dp;
    xz[(size_t)m * (2 * DI) + d] = y;
    __syncthreads();
  }
}

// ---------------------------------------------------------------------------
extern "C" void kernel_launch(void* const* d_in, const int* in_sizes, int n_in,
                              void* d_out, int out_size, void* d_ws, size_t ws_size,
                              hipStream_t stream) {
  const float* x      = (const float*)d_in[0];
  const float* W_in   = (const float*)d_in[1];
  const float* W_out  = (const float*)d_in[2];
  const float* conv_w = (const float*)d_in[3];
  const float* conv_b = (const float*)d_in[4];
  const float* W_x    = (const float*)d_in[5];
  const float* W_dt   = (const float*)d_in[6];
  const float* b_dt   = (const float*)d_in[7];
  const float* A_log  = (const float*)d_in[8];
  const float* D_par  = (const float*)d_in[9];
  float* out = (float*)d_out;

  // Workspace layout (floats): xz | u | params | delta
  float* xz     = (float*)d_ws;                         // MROWS * 2*DI
  float* u      = xz + (size_t)MROWS * 2 * DI;          // MROWS * DI
  float* params = u + (size_t)MROWS * DI;               // MROWS * NP
  float* delta  = params + (size_t)MROWS * NP;          // MROWS * DI

  const dim3 blk(256);

  // 1) xz = x @ W_in^T, with silu fused onto the z half (cols >= DI).
  //    Conv reads raw x_in (cols < DI) which is untouched by the epilogue.
  gemm_wmma_bf16<<<dim3(MROWS / 32, (2 * DI) / 256), blk, 0, stream>>>(
      x, DM, W_in, DM, xz, 2 * DI, 2 * DI, DM,
      EPI_SILU_GE, DI, nullptr);

  // 2) u = silu(causal_conv(x_in) + b)
  conv_silu_k<<<(MROWS * DI) / 256, blk, 0, stream>>>(xz, conv_w, conv_b, u);

  // 3) params = u @ W_x^T       (N=96, K=2048)
  gemm_wmma_bf16<<<dim3(MROWS / 32, 1), blk, 0, stream>>>(
      u, DI, W_x, DI, params, NP, NP, DI,
      EPI_NONE, 0, nullptr);

  // 4) delta = softplus(params[:, :64] @ W_dt^T + 2*b_dt)   (N=2048, K=64)
  gemm_wmma_bf16<<<dim3(MROWS / 32, DI / 256), blk, 0, stream>>>(
      params, NP, W_dt, DTR, delta, DI, DI, DTR,
      EPI_SOFTPLUS2, 0, b_dt);

  // 5) selective scan; y_ssm + u*D written over x_in half of xz
  scan_k<<<dim3(BSZ, DI / 256), blk, 0, stream>>>(delta, u, params, A_log, D_par, xz);

  // 6) out = [y_ssm | silu(z)] @ W_out^T   (N=1024, K=4096)
  gemm_wmma_bf16<<<dim3(MROWS / 32, DM / 256), blk, 0, stream>>>(
      xz, 2 * DI, W_out, 2 * DI, out, DM, DM, 2 * DI,
      EPI_NONE, 0, nullptr);
}
